// HGT_69286412419105
// MI455X (gfx1250) — compile-verified, matrix operand order checked by reading, not run on hardware
//
#include <hip/hip_runtime.h>
#include <hip/hip_bf16.h>
#include <math.h>

// ---------------- constants (match reference) ----------------
#define HID 256
#define NH  8
#define DH  32
#define NLAY 2
#define NV  20000
#define NT_ 50000
#define NE  100000
#define NC  8
#define FIN 64
#define KC  32   // k-chunk staged per TDM transfer

typedef __attribute__((ext_vector_type(16))) __bf16 v16bf;
typedef __attribute__((ext_vector_type(8)))  float  v8f;
typedef __attribute__((ext_vector_type(4)))  unsigned int u32x4;
typedef __attribute__((ext_vector_type(8)))  int i32x8;
typedef __attribute__((ext_vector_type(4)))  int i32x4;

__device__ __forceinline__ v8f wmma_bf16(v16bf a, v16bf b, v8f c) {
  // D = A(16x32) * B(32x16) + C ; f32 accumulate
  return __builtin_amdgcn_wmma_f32_16x16x32_bf16(false, a, false, b, (short)0, c, false, false);
}

__device__ __forceinline__ void atomicMaxF(float* addr, float v) {
  if (v >= 0.0f) atomicMax((int*)addr, __float_as_int(v));
  else           atomicMin((unsigned int*)addr, __float_as_uint(v));
}

// ---- Tensor Data Mover: DMA a (256 rows x KC cols, bf16, row stride K) panel into LDS ----
// D# packing per CDNA5 ISA section 8 (group0: count/lds/global/type, group1: dims/strides).
__device__ __forceinline__ void tdm_load_panel(const __bf16* gsrc, unsigned ldsOff, int K_) {
  unsigned long long ga = (unsigned long long)gsrc;
  u32x4 g0;
  g0[0] = 1u;                                            // count=1, user descriptor
  g0[1] = ldsOff;                                        // lds_addr (bytes)
  g0[2] = (unsigned)ga;                                  // global_addr[31:0]
  g0[3] = (unsigned)((ga >> 32) & 0x1FFFFFFu) | (2u << 30); // global_addr[56:32], type=2
  const unsigned td0 = KC, td1 = 256;                    // tensor dims (tile == tensor window)
  const unsigned long long s0 = (unsigned long long)K_;  // dim0 stride (elements)
  i32x8 g1;
  g1[0] = 0x00010000;                                    // data_size=1 -> 2 bytes/elem
  g1[1] = (int)((td0 & 0xFFFFu) << 16);                  // tensor_dim0[15:0] @ bits 63:48
  g1[2] = (int)(((td0 >> 16) & 0xFFFFu) | ((td1 & 0xFFFFu) << 16));
  g1[3] = (int)(((td1 >> 16) & 0xFFFFu) | ((KC & 0xFFFFu) << 16));  // tile_dim0
  g1[4] = (int)(256u & 0xFFFFu);                         // tile_dim1 (tile_dim2 = 0)
  g1[5] = (int)(s0 & 0xFFFFFFFFu);                       // tensor_dim0_stride lo
  g1[6] = (int)((s0 >> 32) & 0xFFFFu);
  g1[7] = 0;
  i32x4 g2 = {0, 0, 0, 0}, g3 = {0, 0, 0, 0};
#if __clang_major__ >= 23
  i32x8 g4 = {0, 0, 0, 0, 0, 0, 0, 0};
  __builtin_amdgcn_tensor_load_to_lds(g0, g1, g2, g3, g4, 0);
#else
  __builtin_amdgcn_tensor_load_to_lds(g0, g1, g2, g3, 0);
#endif
}

// ---------------- elementwise helpers ----------------
__global__ void fill_f32_kernel(float* p, float v, long n) {
  long i = (long)blockIdx.x * blockDim.x + threadIdx.x;
  if (i < n) p[i] = v;
}

__global__ void cast_bf16_kernel(const float* __restrict__ in, __bf16* __restrict__ out, long n) {
  long i = (long)blockIdx.x * blockDim.x + threadIdx.x;
  if (i < n) out[i] = (__bf16)in[i];
}

__global__ void gelu_cast_kernel(const float* __restrict__ in, __bf16* __restrict__ out, long n) {
  long i = (long)blockIdx.x * blockDim.x + threadIdx.x;
  if (i < n) {
    float x = in[i];
    float g = 0.5f * x * (1.0f + erff(x * 0.70710678118654752f));
    out[i] = (__bf16)g;
  }
}

// dst[m*K + k] = (bf16) src[k*M + m]   (store W^T so B fragments are contiguous)
__global__ void transpose_cast_kernel(const float* __restrict__ src, __bf16* __restrict__ dst,
                                      int K, int M) {
  long i = (long)blockIdx.x * blockDim.x + threadIdx.x;
  if (i >= (long)K * M) return;
  int m = (int)(i / K), k = (int)(i % K);
  dst[i] = (__bf16)src[(long)k * M + m];
}

// a_relT[g][f][d] = a_rel[g][d][f],  g = (l*3+r)*NH + h
__global__ void rel_transpose_cast_kernel(const float* __restrict__ src, __bf16* __restrict__ dst,
                                          long total) {
  long i = (long)blockIdx.x * blockDim.x + threadIdx.x;
  if (i >= total) return;
  int d = (int)(i & 31);
  int f = (int)((i >> 5) & 31);
  long g = i >> 10;
  dst[i] = (__bf16)src[(g * 32 + d) * 32 + f];
}

// ---------------- WMMA GEMM:  Out(N x 256) = act(A(N x K) @ W + bias) ----------------
// A: bf16 row-major; Bt: bf16 (256 x K) = W^T row-major.
// B panels are TDM-DMA'd into double-buffered LDS; A fragments are register-prefetched.
// mode 0: relu ; mode 1: none ; mode 2: beta*(acc+bias)+(1-beta)*hold, beta=sigmoid(*skip)
__global__ void gemm_bf16_kernel(const __bf16* __restrict__ A, const __bf16* __restrict__ Bt,
                                 const float* __restrict__ bias, const float* __restrict__ hold,
                                 const float* __restrict__ skip, float* __restrict__ Out,
                                 int N, int K, int mode) {
  __shared__ __bf16 sB[2][256 * KC];  // 2 x 16KB double buffer
  const int wave = threadIdx.x >> 5;
  const int lane = threadIdx.x & 31;
  const int rowTile = blockIdx.x * 32 + (wave >> 2) * 16;  // 2 row tiles per block
  const int colGrp = (wave & 3) * 64;                      // 4 col groups of 64

  int aRow = rowTile + (lane & 15);
  if (aRow >= N) aRow = N - 1;                             // clamp; stores are guarded
  const int kHalf = (lane >> 4) * 16;
  const __bf16* aPtr = A + (size_t)aRow * K + kHalf;

  const unsigned lds0 = (unsigned)(unsigned long long)(void*)&sB[0][0];
  const unsigned lds1 = (unsigned)(unsigned long long)(void*)&sB[1][0];
  const int nCh = K / KC;

  if (wave == 0) {
    tdm_load_panel(Bt, lds0, K);
    __builtin_amdgcn_s_wait_tensorcnt(0);
  }
  v16bf aNext = *(const v16bf*)aPtr;
  __syncthreads();

  v8f acc[4] = {v8f{}, v8f{}, v8f{}, v8f{}};
  for (int c = 0; c < nCh; ++c) {
    if (wave == 0 && c + 1 < nCh)
      tdm_load_panel(Bt + (size_t)(c + 1) * KC, ((c + 1) & 1) ? lds1 : lds0, K);
    v16bf afrag = aNext;
    if (c + 1 < nCh) aNext = *(const v16bf*)(aPtr + (size_t)(c + 1) * KC);
    const __bf16* sb = sB[c & 1];
    // hoist all B-fragment LDS loads so they clause together (one dscnt wait,
    // then 4 back-to-back WMMAs co-executing while next chunk streams in)
    v16bf bfrag[4];
#pragma unroll
    for (int s = 0; s < 4; ++s)
      bfrag[s] = *(const v16bf*)(sb + (size_t)(colGrp + s * 16 + (lane & 15)) * KC + kHalf);
#pragma unroll
    for (int s = 0; s < 4; ++s)
      acc[s] = wmma_bf16(afrag, bfrag[s], acc[s]);
    if (wave == 0 && c + 1 < nCh) __builtin_amdgcn_s_wait_tensorcnt(0);
    __syncthreads();
  }

  float beta = 1.0f, invb = 0.0f;
  if (mode == 2) {
    float sv = skip[0];
    beta = 1.0f / (1.0f + __expf(-sv));
    invb = 1.0f - beta;
  }
  const int rOff = (lane >> 4) * 8;   // C layout: VGPR r -> m = r + 8*(lane/16), n = lane%16
  const int nCol = lane & 15;
#pragma unroll
  for (int s = 0; s < 4; ++s) {
    int col = colGrp + s * 16 + nCol;
    float b = bias ? bias[col] : 0.0f;
#pragma unroll
    for (int r = 0; r < 8; ++r) {
      int row = rowTile + rOff + r;
      if (row < N) {
        float v = acc[s][r] + b;
        if (mode == 0)      v = fmaxf(v, 0.0f);
        else if (mode == 2) v = beta * v + invb * hold[(size_t)row * HID + col];
        Out[(size_t)row * HID + col] = v;
      }
    }
  }
}

// ---------------- per-head rel apply:  Out[n,h,f] = sum_d X[n,h,d] * R[h,d,f] ----------------
// RT: bf16 [NH][32 f][32 d] (pre-transposed). One wave per (16-node tile, head).
__global__ void rel_apply_kernel(const float* __restrict__ X, const __bf16* __restrict__ RT,
                                 float* __restrict__ Out, int N) {
  const int h = threadIdx.x >> 5;          // 8 waves = 8 heads
  const int lane = threadIdx.x & 31;
  const int rowTile = blockIdx.x * 16;
  int aRow = rowTile + (lane & 15);
  if (aRow >= N) aRow = N - 1;
  const int kHalf = (lane >> 4) * 16;      // d offset for this lane half

  const float4* xp = (const float4*)(X + (size_t)aRow * HID + h * DH + kHalf);
  float4 x0 = xp[0], x1 = xp[1], x2 = xp[2], x3 = xp[3];
  v16bf a;
  a[0]=(__bf16)x0.x; a[1]=(__bf16)x0.y; a[2]=(__bf16)x0.z; a[3]=(__bf16)x0.w;
  a[4]=(__bf16)x1.x; a[5]=(__bf16)x1.y; a[6]=(__bf16)x1.z; a[7]=(__bf16)x1.w;
  a[8]=(__bf16)x2.x; a[9]=(__bf16)x2.y; a[10]=(__bf16)x2.z; a[11]=(__bf16)x2.w;
  a[12]=(__bf16)x3.x; a[13]=(__bf16)x3.y; a[14]=(__bf16)x3.z; a[15]=(__bf16)x3.w;

  const __bf16* rbase = RT + (size_t)h * DH * DH;
  v16bf b0 = *(const v16bf*)(rbase + (size_t)(lane & 15) * DH + kHalf);
  v16bf b1 = *(const v16bf*)(rbase + (size_t)(16 + (lane & 15)) * DH + kHalf);
  v8f c0 = v8f{}, c1 = v8f{};
  c0 = wmma_bf16(a, b0, c0);
  c1 = wmma_bf16(a, b1, c1);

  const int rOff = (lane >> 4) * 8;
  const int nCol = lane & 15;
#pragma unroll
  for (int r = 0; r < 8; ++r) {
    int row = rowTile + rOff + r;
    if (row < N) {
      Out[(size_t)row * HID + h * DH + nCol]      = c0[r];
      Out[(size_t)row * HID + h * DH + 16 + nCol] = c1[r];
    }
  }
}

// ---------------- edge passes ----------------
__global__ void edge_logit_kernel(const int* __restrict__ si, const int* __restrict__ di,
                                  const float* __restrict__ q, const float* __restrict__ ka,
                                  const float* __restrict__ prel, float* __restrict__ logit,
                                  float* __restrict__ mbuf, float scale, int nE) {
  int idx = blockIdx.x * blockDim.x + threadIdx.x;
  if (idx >= nE * NH) return;
  int e = idx >> 3, h = idx & 7;
  int s = si[e], d = di[e];
  const float4* qp = (const float4*)(q + (size_t)d * HID + h * DH);
  const float4* kp = (const float4*)(ka + (size_t)s * HID + h * DH);
  float dot = 0.0f;
#pragma unroll
  for (int i = 0; i < 8; ++i) {
    float4 av = qp[i], bv = kp[i];
    dot += av.x * bv.x + av.y * bv.y + av.z * bv.z + av.w * bv.w;
  }
  float lg = dot * prel[h] * scale;
  logit[idx] = lg;
  atomicMaxF(&mbuf[(size_t)d * NH + h], lg);
}

__global__ void edge_exp_kernel(const int* __restrict__ di, float* __restrict__ logit,
                                const float* __restrict__ mbuf, float* __restrict__ sbuf, int nE) {
  int idx = blockIdx.x * blockDim.x + threadIdx.x;
  if (idx >= nE * NH) return;
  int e = idx >> 3, h = idx & 7;
  int d = di[e];
  float ex = __expf(logit[idx] - mbuf[(size_t)d * NH + h]);
  logit[idx] = ex;
  atomicAdd(&sbuf[(size_t)d * NH + h], ex);
}

__global__ void edge_agg_kernel(const int* __restrict__ si, const int* __restrict__ di,
                                const float* __restrict__ logit, const float* __restrict__ sbuf,
                                const float* __restrict__ vm, float* __restrict__ agg, int nE) {
  int idx = blockIdx.x * blockDim.x + threadIdx.x;
  if (idx >= nE * NH) return;
  int e = idx >> 3, h = idx & 7;
  int s = si[e], d = di[e];
  float w = logit[idx] / (sbuf[(size_t)d * NH + h] + 1e-16f);
  const float* vp = vm + (size_t)s * HID + h * DH;
  float* ap = agg + (size_t)d * HID + h * DH;
#pragma unroll
  for (int f = 0; f < DH; ++f) atomicAdd(&ap[f], w * vp[f]);
}

// ---------------- final scorer ----------------
__global__ void vpart_kernel(const float* __restrict__ hv, const int* __restrict__ current,
                             const float* __restrict__ Ws1, const float* __restrict__ bs1,
                             float* __restrict__ vpart) {
  int idx = blockIdx.x * blockDim.x + threadIdx.x;
  if (idx >= NC * HID) return;
  int c = idx / HID, m = idx % HID;
  const float* row = hv + (size_t)current[c * 2] * HID;
  float acc = bs1[m];
  for (int k = 0; k < HID; ++k) acc += row[k] * Ws1[(size_t)(HID + k) * HID + m];
  vpart[idx] = acc;
}

__global__ void final_kernel(const float* __restrict__ tpart, const float* __restrict__ vpart,
                             const float* __restrict__ Ws2, const float* __restrict__ bs2,
                             float* __restrict__ out0, float* __restrict__ out1) {
  int t = blockIdx.x;                 // one t-node per block, 8 waves = 8 candidates
  int c = threadIdx.x >> 5;
  int lane = threadIdx.x & 31;
  const float* tp = tpart + (size_t)t * HID;
  const float* vp = vpart + (size_t)c * HID;
  float a0 = 0.0f, a1 = 0.0f;
  for (int i = lane; i < HID; i += 32) {
    float hmid = fmaxf(tp[i] + vp[i], 0.0f);
    a0 += hmid * Ws2[i * 2 + 0];
    a1 += hmid * Ws2[i * 2 + 1];
  }
#pragma unroll
  for (int off = 16; off; off >>= 1) { a0 += __shfl_xor(a0, off, 32); a1 += __shfl_xor(a1, off, 32); }
  if (lane == 0) {
    out0[(size_t)c * NT_ + t] = a0 + bs2[0];
    float z = a1 + bs2[1];
    out1[(size_t)c * NT_ + t] = 1.0f / (1.0f + __expf(-z));
  }
}

// ---------------- host orchestration ----------------
struct Bump {
  char* base; size_t off;
  template <typename T> T* get(size_t n) {
    T* p = (T*)(base + off);
    off = (off + n * sizeof(T) + 255) & ~(size_t)255;
    return p;
  }
};

extern "C" void kernel_launch(void* const* d_in, const int* in_sizes, int n_in,
                              void* d_out, int out_size, void* d_ws, size_t ws_size,
                              hipStream_t stream) {
  (void)in_sizes; (void)n_in; (void)out_size; (void)ws_size;
  const float* x_v = (const float*)d_in[0];
  const float* x_t = (const float*)d_in[1];
  const int* si_[3] = {(const int*)d_in[2], (const int*)d_in[4], (const int*)d_in[6]};
  const int* di_[3] = {(const int*)d_in[3], (const int*)d_in[5], (const int*)d_in[7]};
  const int* current = (const int*)d_in[8];
  const float* x_unused = nullptr; (void)x_unused;
  const float* W_in_v = (const float*)d_in[9];  const float* b_in_v = (const float*)d_in[10];
  const float* W_in_t = (const float*)d_in[11]; const float* b_in_t = (const float*)d_in[12];
  const float* Wk = (const float*)d_in[13]; const float* bk = (const float*)d_in[14];
  const float* Wq = (const float*)d_in[15]; const float* bq = (const float*)d_in[16];
  const float* Wv = (const float*)d_in[17]; const float* bv = (const float*)d_in[18];
  const float* Wa = (const float*)d_in[19]; const float* ba = (const float*)d_in[20];
  const float* skip = (const float*)d_in[21];
  const float* a_rel = (const float*)d_in[22];
  const float* m_rel = (const float*)d_in[23];
  const float* p_rel = (const float*)d_in[24];
  const float* Ws1 = (const float*)d_in[25]; const float* bs1 = (const float*)d_in[26];
  const float* Ws2 = (const float*)d_in[27]; const float* bs2 = (const float*)d_in[28];

  Bump ws{(char*)d_ws, 0};
  float* hbuf[2]; hbuf[0] = ws.get<float>((size_t)NV * HID);  hbuf[1] = ws.get<float>((size_t)NT_ * HID);
  __bf16* hbf[2]; hbf[0] = ws.get<__bf16>((size_t)NV * HID);  hbf[1] = ws.get<__bf16>((size_t)NT_ * HID);
  __bf16* xbf   = ws.get<__bf16>((size_t)NT_ * FIN);
  float* kbuf   = ws.get<float>((size_t)NT_ * HID);
  float* qbuf   = ws.get<float>((size_t)NT_ * HID);
  float* vbuf   = ws.get<float>((size_t)NT_ * HID);
  float* kabuf  = ws.get<float>((size_t)NT_ * HID);
  float* vmbuf  = ws.get<float>((size_t)NT_ * HID);
  float* aggb[2]; aggb[0] = ws.get<float>((size_t)NV * HID); aggb[1] = ws.get<float>((size_t)NT_ * HID);
  __bf16* gbf   = ws.get<__bf16>((size_t)NT_ * HID);
  float* logit  = ws.get<float>((size_t)NE * NH);
  float* mbuf   = ws.get<float>((size_t)NT_ * NH);
  float* sbuf   = ws.get<float>((size_t)NT_ * NH);
  __bf16* WinvT = ws.get<__bf16>((size_t)HID * FIN);
  __bf16* WintT = ws.get<__bf16>((size_t)HID * FIN);
  __bf16* WkT   = ws.get<__bf16>((size_t)NLAY * 2 * HID * HID);
  __bf16* WqT   = ws.get<__bf16>((size_t)NLAY * 2 * HID * HID);
  __bf16* WvT   = ws.get<__bf16>((size_t)NLAY * 2 * HID * HID);
  __bf16* WaT   = ws.get<__bf16>((size_t)NLAY * 2 * HID * HID);
  __bf16* Ws1T  = ws.get<__bf16>((size_t)HID * HID);
  __bf16* arelT = ws.get<__bf16>((size_t)NLAY * 3 * NH * DH * DH);
  __bf16* mrelT = ws.get<__bf16>((size_t)NLAY * 3 * NH * DH * DH);
  float* vpart  = ws.get<float>((size_t)NC * HID);

  auto blocks = [](long n) { return dim3((unsigned)((n + 255) / 256)); };
  auto gemm = [&](const __bf16* A, const __bf16* Bt, const float* bias, const float* hold,
                  const float* sk, float* out, int N, int K, int mode) {
    gemm_bf16_kernel<<<dim3((N + 31) / 32), dim3(256), 0, stream>>>(A, Bt, bias, hold, sk, out, N, K, mode);
  };

  // ---- weight prep (bf16, transposed) ----
  transpose_cast_kernel<<<blocks((long)FIN * HID), 256, 0, stream>>>(W_in_v, WinvT, FIN, HID);
  transpose_cast_kernel<<<blocks((long)FIN * HID), 256, 0, stream>>>(W_in_t, WintT, FIN, HID);
  for (int i = 0; i < NLAY * 2; ++i) {
    size_t o = (size_t)i * HID * HID;
    transpose_cast_kernel<<<blocks((long)HID * HID), 256, 0, stream>>>(Wk + o, WkT + o, HID, HID);
    transpose_cast_kernel<<<blocks((long)HID * HID), 256, 0, stream>>>(Wq + o, WqT + o, HID, HID);
    transpose_cast_kernel<<<blocks((long)HID * HID), 256, 0, stream>>>(Wv + o, WvT + o, HID, HID);
    transpose_cast_kernel<<<blocks((long)HID * HID), 256, 0, stream>>>(Wa + o, WaT + o, HID, HID);
  }
  transpose_cast_kernel<<<blocks((long)HID * HID), 256, 0, stream>>>(Ws1, Ws1T, HID, HID); // top half
  long relN = (long)NLAY * 3 * NH * DH * DH;
  rel_transpose_cast_kernel<<<blocks(relN), 256, 0, stream>>>(a_rel, arelT, relN);
  rel_transpose_cast_kernel<<<blocks(relN), 256, 0, stream>>>(m_rel, mrelT, relN);

  // ---- input projections: h = relu(x @ W_in + b) ----
  cast_bf16_kernel<<<blocks((long)NV * FIN), 256, 0, stream>>>(x_v, xbf, (long)NV * FIN);
  gemm(xbf, WinvT, b_in_v, nullptr, nullptr, hbuf[0], NV, FIN, 0);
  cast_bf16_kernel<<<blocks((long)NT_ * FIN), 256, 0, stream>>>(x_t, xbf, (long)NT_ * FIN);
  gemm(xbf, WintT, b_in_t, nullptr, nullptr, hbuf[1], NT_, FIN, 0);

  const int stA[3] = {0, 1, 1}, dtA[3] = {1, 0, 1};
  const int nNode[2] = {NV, NT_};
  const float scale = 0.17677669529663687f; // 1/sqrt(32)

  for (int l = 0; l < NLAY; ++l) {
    cast_bf16_kernel<<<blocks((long)NV * HID), 256, 0, stream>>>(hbuf[0], hbf[0], (long)NV * HID);
    cast_bf16_kernel<<<blocks((long)NT_ * HID), 256, 0, stream>>>(hbuf[1], hbf[1], (long)NT_ * HID);
    fill_f32_kernel<<<blocks((long)NV * HID), 256, 0, stream>>>(aggb[0], 0.0f, (long)NV * HID);
    fill_f32_kernel<<<blocks((long)NT_ * HID), 256, 0, stream>>>(aggb[1], 0.0f, (long)NT_ * HID);

    for (int r = 0; r < 3; ++r) {
      int st = stA[r], dt = dtA[r];
      int nS = nNode[st], nD = nNode[dt];
      size_t wS = ((size_t)l * 2 + st) * HID * HID, wD = ((size_t)l * 2 + dt) * HID * HID;
      size_t bS = ((size_t)l * 2 + st) * HID,       bD = ((size_t)l * 2 + dt) * HID;
      size_t ro = ((size_t)l * 3 + r) * NH * DH * DH;

      gemm(hbf[st], WkT + wS, bk + bS, nullptr, nullptr, kbuf, nS, HID, 1);
      rel_apply_kernel<<<dim3((nS + 15) / 16), dim3(256), 0, stream>>>(kbuf, arelT + ro, kabuf, nS);
      gemm(hbf[dt], WqT + wD, bq + bD, nullptr, nullptr, qbuf, nD, HID, 1);
      gemm(hbf[st], WvT + wS, bv + bS, nullptr, nullptr, vbuf, nS, HID, 1);
      rel_apply_kernel<<<dim3((nS + 15) / 16), dim3(256), 0, stream>>>(vbuf, mrelT + ro, vmbuf, nS);

      fill_f32_kernel<<<blocks((long)nD * NH), 256, 0, stream>>>(mbuf, -INFINITY, (long)nD * NH);
      fill_f32_kernel<<<blocks((long)nD * NH), 256, 0, stream>>>(sbuf, 0.0f, (long)nD * NH);

      long nEH = (long)NE * NH;
      edge_logit_kernel<<<blocks(nEH), 256, 0, stream>>>(si_[r], di_[r], qbuf, kabuf,
                                                         p_rel + ((size_t)l * 3 + r) * NH,
                                                         logit, mbuf, scale, NE);
      edge_exp_kernel<<<blocks(nEH), 256, 0, stream>>>(di_[r], logit, mbuf, sbuf, NE);
      edge_agg_kernel<<<blocks(nEH), 256, 0, stream>>>(si_[r], di_[r], logit, sbuf, vmbuf, aggb[dt], NE);
    }

    for (int t = 0; t < 2; ++t) {
      long n = (long)nNode[t] * HID;
      size_t w = ((size_t)l * 2 + t) * HID * HID, b = ((size_t)l * 2 + t) * HID;
      gelu_cast_kernel<<<blocks(n), 256, 0, stream>>>(aggb[t], gbf, n);
      gemm(gbf, WaT + w, ba + b, hbuf[t], skip + (size_t)l * 2 + t, hbuf[t], nNode[t], HID, 2);
    }
  }

  // ---- final scorer ----
  cast_bf16_kernel<<<blocks((long)NT_ * HID), 256, 0, stream>>>(hbuf[1], hbf[1], (long)NT_ * HID);
  float* tpart = kabuf; // free after layers
  gemm(hbf[1], Ws1T, nullptr, nullptr, nullptr, tpart, NT_, HID, 1);
  vpart_kernel<<<blocks((long)NC * HID), 256, 0, stream>>>(hbuf[0], current, Ws1, bs1, vpart);
  float* out0 = (float*)d_out;
  float* out1 = out0 + (size_t)NC * NT_;
  final_kernel<<<dim3(NT_), dim3(256), 0, stream>>>(tpart, vpart, Ws2, bs2, out0, out1);
}
